// SinkhornDistance_25305947308135
// MI455X (gfx1250) — matrix operand order repaired and checked by reading, not compiled, not run on hardware
//
#include <hip/hip_runtime.h>
#include <hip/hip_bf16.h>

// ---------------- problem constants ----------------
#define NB 65536     // batch rows
#define NK 256       // prototypes / columns
#define ND 128       // feature dim
#define NITER 50
#define EPSI 0.1f
#define NEG_INV_EPSI (-10.0f)   // -1/EPSI, avoid runtime divide
#define LOGEPS 1e-8f

#define CPB 512      // column-pass stage-1 blocks (128 rows each)
#define ROWS_PER_CPB (NB / CPB)   // 128

// ---------------- workspace layout (bytes) ----------------
#define OFF_LOGK  ((size_t)0)                         // 65536*256*4 = 64 MB
#define OFF_XNH   (OFF_LOGK + (size_t)NB*NK*4)        // 65536*128*2 = 16 MB
#define OFF_YNH   (OFF_XNH  + (size_t)NB*ND*2)        // 256*128*2   = 64 KB
#define OFF_U     (OFF_YNH  + (size_t)NK*ND*2)        // 65536*4
#define OFF_V     (OFF_U    + (size_t)NB*4)           // 256*4 -> pad 1KB
#define OFF_CPM   (OFF_V    + (size_t)1024)           // 512*256*4
#define OFF_CPS   (OFF_CPM  + (size_t)CPB*NK*4)       // 512*256*4
#define OFF_RP    (OFF_CPS  + (size_t)CPB*NK*4)       // 8192*4

typedef __attribute__((ext_vector_type(16))) _Float16 v16h;
typedef __attribute__((ext_vector_type(8)))  float    v8f;
typedef __attribute__((ext_vector_type(2)))  _Float16 h2t;
typedef __attribute__((ext_vector_type(4)))  _Float16 h4t;

__device__ __forceinline__ float wave_max(float v) {
  #pragma unroll
  for (int m = 16; m > 0; m >>= 1) v = fmaxf(v, __shfl_xor(v, m, 32));
  return v;
}
__device__ __forceinline__ float wave_sum(float v) {
  #pragma unroll
  for (int m = 16; m > 0; m >>= 1) v += __shfl_xor(v, m, 32);
  return v;
}

// ---------- 1) L2-normalize rows, emit f16 ----------
__global__ void rownorm_f16_kernel(const float* __restrict__ in,
                                   _Float16* __restrict__ out, int nrows) {
  int wave = threadIdx.x >> 5, lane = threadIdx.x & 31;
  int row = blockIdx.x * 8 + wave;
  if (row >= nrows) return;   // uniform per wave
  const float4 x = ((const float4*)(in + (size_t)row * ND))[lane];
  float ss = x.x * x.x + x.y * x.y + x.z * x.z + x.w * x.w;
  ss = wave_sum(ss);
  float inv = 1.0f / fmaxf(sqrtf(ss), 1e-12f);
  h4t o;
  o.x = (_Float16)(x.x * inv); o.y = (_Float16)(x.y * inv);
  o.z = (_Float16)(x.z * inv); o.w = (_Float16)(x.w * inv);
  *(h4t*)(out + (size_t)row * ND + lane * 4) = o;
}

// ---------- 2) init v = 0 ----------
__global__ void init_v_kernel(float* __restrict__ v) { v[threadIdx.x] = 0.0f; }

// ---------- 3) WMMA GEMM: logK = log(exp(-10*(1 - xn.yn^T)) + 1e-8) ----------
// grid = (512, 16), block = 256 (8 waves; each wave -> one 16x16 output tile)
__global__ void gemm_logk_kernel(const _Float16* __restrict__ xnh,
                                 const _Float16* __restrict__ ynh,
                                 float* __restrict__ logK) {
  int wave = threadIdx.x >> 5, lane = threadIdx.x & 31;
  int rowTile = (blockIdx.x * 8 + wave) * 16;  // M base
  int colTile = blockIdx.y * 16;               // N base
  int mn   = lane & 15;
  int half = lane >> 4;

  const _Float16* arow = xnh + (size_t)(rowTile + mn) * ND;  // A: row mn of tile
  const _Float16* brow = ynh + (size_t)(colTile + mn) * ND;  // B: proto mn of tile

  v8f acc = {0.f, 0.f, 0.f, 0.f, 0.f, 0.f, 0.f, 0.f};

  #pragma unroll
  for (int kk = 0; kk < ND / 32; kk++) {
    v16h a, b;
    #pragma unroll
    for (int r = 0; r < 8; r++) {
      // A fragment layout (16-bit A 16x32, ISA 7.12.2)
      int ka = kk * 32 + ((r < 4) ? 0 : 16) + half * 8 + (r & 3) * 2;
      h2t ta = *(const h2t*)(arow + ka);
      a[2 * r] = ta.x; a[2 * r + 1] = ta.y;
      // B fragment layout (same K order as A)
      int kb = kk * 32 + ((r < 4) ? 0 : 16) + half * 8 + (r & 3) * 2;
      h2t tb = *(const h2t*)(brow + kb);
      b[2 * r] = tb.x; b[2 * r + 1] = tb.y;
    }
    acc = __builtin_amdgcn_wmma_f32_16x16x32_f16(
        false, a, false, b, (short)0, acc, false, false);
  }

  // C/D layout: lane n = lane&15, VGPR r -> M = (lane>>4)*8 + r
  int n = mn, mbase = half * 8;
  #pragma unroll
  for (int r = 0; r < 8; r++) {
    float cosv = acc[r];
    float Cc = 1.0f - cosv;
    float lk = __logf(__expf(Cc * NEG_INV_EPSI) + LOGEPS);   // multiply, not divide
    logK[(size_t)(rowTile + mbase + r) * NK + colTile + n] = lk;
  }
}

// ---------- 4a) u-update: per-row logsumexp (wave per row) ----------
// lane owns columns [lane*8, lane*8+8): two b128 loads per lane, still coalesced
__global__ void sinkhorn_u_kernel(const float* __restrict__ logK,
                                  const float* __restrict__ v,
                                  float* __restrict__ u) {
  __shared__ float vs[NK];
  vs[threadIdx.x] = v[threadIdx.x];
  __syncthreads();
  int wave = threadIdx.x >> 5, lane = threadIdx.x & 31;
  int row = blockIdx.x * 8 + wave;
  const float4* rowp = (const float4*)(logK + (size_t)row * NK) + lane * 2;
  const float4* vp   = (const float4*)vs + lane * 2;
  float4 l0 = rowp[0], l1 = rowp[1];
  float4 v0 = vp[0],  v1 = vp[1];
  float a[8];
  a[0] = l0.x + v0.x; a[1] = l0.y + v0.y; a[2] = l0.z + v0.z; a[3] = l0.w + v0.w;
  a[4] = l1.x + v1.x; a[5] = l1.y + v1.y; a[6] = l1.z + v1.z; a[7] = l1.w + v1.w;
  float m = a[0];
  #pragma unroll
  for (int t = 1; t < 8; t++) m = fmaxf(m, a[t]);
  m = wave_max(m);
  float s = 0.0f;
  #pragma unroll
  for (int t = 0; t < 8; t++) s += __expf(a[t] - m);
  s = wave_sum(s);
  if (lane == 0) {
    const float log_mu = logf(1.0f / (float)NB + LOGEPS);
    u[row] = log_mu - (m + __logf(s));
  }
}

// ---------- 4b) v-update stage 1: per-block online LSE over 128 rows, all 256 cols ----------
__global__ void sinkhorn_vpart_kernel(const float* __restrict__ logK,
                                      const float* __restrict__ u,
                                      float* __restrict__ cpm,
                                      float* __restrict__ cps) {
  __shared__ float us[ROWS_PER_CPB];
  int tid = threadIdx.x;
  if (tid < ROWS_PER_CPB) us[tid] = u[(size_t)blockIdx.x * ROWS_PER_CPB + tid];
  __syncthreads();
  int j = tid;  // column
  const float* base = logK + (size_t)blockIdx.x * ROWS_PER_CPB * NK + j;
  float m = -3.4e38f, s = 0.0f;
  for (int i = 0; i < ROWS_PER_CPB; i++) {
    float a = base[(size_t)i * NK] + us[i];
    if (a <= m) {
      s += __expf(a - m);
    } else {
      s = s * __expf(m - a) + 1.0f;
      m = a;
    }
  }
  cpm[blockIdx.x * NK + j] = m;
  cps[blockIdx.x * NK + j] = s;
}

// ---------- 4c) v-update stage 2: merge partials ----------
__global__ void sinkhorn_vmerge_kernel(const float* __restrict__ cpm,
                                       const float* __restrict__ cps,
                                       float* __restrict__ v) {
  int j = threadIdx.x;
  float M = -3.4e38f, S = 0.0f;
  for (int b = 0; b < CPB; b++) {
    float m = cpm[b * NK + j];
    float s = cps[b * NK + j];
    if (m <= M) {
      S += s * __expf(m - M);
    } else {
      S = S * __expf(M - m) + s;
      M = m;
    }
  }
  const float log_nu = logf(1.0f / (float)NK + LOGEPS);
  v[j] = log_nu - (M + __logf(S));
}

// ---------- 5) final: sum(pi * C * softmax(|coord|)) -> per-block partials ----------
__global__ void final_part_kernel(const float* __restrict__ logK,
                                  const float* __restrict__ u,
                                  const float* __restrict__ v,
                                  const float* __restrict__ coords,
                                  float* __restrict__ part) {
  __shared__ float vs[NK];
  __shared__ float wacc[8];
  vs[threadIdx.x] = v[threadIdx.x];
  __syncthreads();
  int wave = threadIdx.x >> 5, lane = threadIdx.x & 31;
  int row = blockIdx.x * 8 + wave;
  const float4* lkr = (const float4*)(logK + (size_t)row * NK) + lane * 2;
  const float4* crr = (const float4*)(coords + (size_t)row * NK) + lane * 2;
  const float4* vp  = (const float4*)vs + lane * 2;
  float ui = u[row];

  float4 k0 = lkr[0], k1 = lkr[1];
  float4 c0 = crr[0], c1 = crr[1];
  float4 v0 = vp[0],  v1 = vp[1];

  float lk[8] = {k0.x, k0.y, k0.z, k0.w, k1.x, k1.y, k1.z, k1.w};
  float cw[8] = {fabsf(c0.x), fabsf(c0.y), fabsf(c0.z), fabsf(c0.w),
                 fabsf(c1.x), fabsf(c1.y), fabsf(c1.z), fabsf(c1.w)};
  float vv[8] = {v0.x, v0.y, v0.z, v0.w, v1.x, v1.y, v1.z, v1.w};

  float cmax = cw[0];
  #pragma unroll
  for (int t = 1; t < 8; t++) cmax = fmaxf(cmax, cw[t]);
  cmax = wave_max(cmax);
  float csum = 0.0f;
  #pragma unroll
  for (int t = 0; t < 8; t++) csum += __expf(cw[t] - cmax);
  csum = wave_sum(csum);
  float inv_csum = 1.0f / csum;

  float acc = 0.0f;
  #pragma unroll
  for (int t = 0; t < 8; t++) {
    float Km = fmaxf(__expf(lk[t]) - LOGEPS, 1e-20f);
    float Cc = -EPSI * __logf(Km);                 // recover cost from logK
    float pi = __expf(ui + lk[t] + vv[t]);
    float w  = __expf(cw[t] - cmax) * inv_csum;
    acc += pi * Cc * w;
  }
  acc = wave_sum(acc);
  if (lane == 0) wacc[wave] = acc;
  __syncthreads();
  if (threadIdx.x == 0) {
    float s = 0.0f;
    #pragma unroll
    for (int w = 0; w < 8; w++) s += wacc[w];   // fixed order: deterministic
    part[blockIdx.x] = s;
  }
}

// ---------- 6) deterministic final reduce of 8192 partials ----------
__global__ void final_reduce_kernel(const float* __restrict__ part,
                                    float* __restrict__ out) {
  __shared__ float sm[256];
  float s = 0.0f;
  for (int k = 0; k < 32; k++) s += part[threadIdx.x + 256 * k];
  sm[threadIdx.x] = s;
  __syncthreads();
  for (int off = 128; off > 0; off >>= 1) {
    if (threadIdx.x < off) sm[threadIdx.x] += sm[threadIdx.x + off];
    __syncthreads();
  }
  if (threadIdx.x == 0) out[0] = sm[0];
}

extern "C" void kernel_launch(void* const* d_in, const int* in_sizes, int n_in,
                              void* d_out, int out_size, void* d_ws, size_t ws_size,
                              hipStream_t stream) {
  const float* x      = (const float*)d_in[0];  // (B, D) f32
  const float* protos = (const float*)d_in[1];  // (K, D) f32
  const float* coords = (const float*)d_in[2];  // (B, K) f32

  char* ws = (char*)d_ws;
  float*    logK = (float*)(ws + OFF_LOGK);
  _Float16* xnh  = (_Float16*)(ws + OFF_XNH);
  _Float16* ynh  = (_Float16*)(ws + OFF_YNH);
  float*    u    = (float*)(ws + OFF_U);
  float*    v    = (float*)(ws + OFF_V);
  float*    cpm  = (float*)(ws + OFF_CPM);
  float*    cps  = (float*)(ws + OFF_CPS);
  float*    part = (float*)(ws + OFF_RP);
  float*    out  = (float*)d_out;

  // normalize
  rownorm_f16_kernel<<<NB / 8, 256, 0, stream>>>(x, xnh, NB);
  rownorm_f16_kernel<<<NK / 8, 256, 0, stream>>>(protos, ynh, NK);
  init_v_kernel<<<1, NK, 0, stream>>>(v);

  // cost matrix -> logK (WMMA f16 -> f32 acc), L2-resident afterwards (64 MB)
  gemm_logk_kernel<<<dim3(NB / (8 * 16), NK / 16), 256, 0, stream>>>(xnh, ynh, logK);

  // 50 Sinkhorn iterations (row LSE, then 2-stage column LSE)
  for (int it = 0; it < NITER; it++) {
    sinkhorn_u_kernel<<<NB / 8, 256, 0, stream>>>(logK, v, u);
    sinkhorn_vpart_kernel<<<CPB, NK, 0, stream>>>(logK, u, cpm, cps);
    sinkhorn_vmerge_kernel<<<1, NK, 0, stream>>>(cpm, cps, v);
  }

  // fused transport-plan * cost * coordinate-softmax sum
  final_part_kernel<<<NB / 8, 256, 0, stream>>>(logK, u, v, coords, part);
  final_reduce_kernel<<<1, 256, 0, stream>>>(part, out);
}
